// CRFDecode_9363028706247
// MI455X (gfx1250) — matrix operand order repaired and checked by reading, not compiled, not run on hardware
//
#include <hip/hip_runtime.h>
#include <hip/hip_bf16.h>

// ---------------------------------------------------------------------------
// CRF decode + NLL for MI455X (gfx1250, wave32).
//
// log-partition forward: exp-domain reformulation ->
//   alpha'[b][j] = m_b + log( sum_i exp(alpha[b][i]-m_b) * exp(W[i][j]) ) + logit
// Inner sum = [16x32]x[32x32] matmul per step -> V_WMMA_F32_16X16X32_BF16,
// bf16 hi/lo compensated (3 WMMAs per 16x16 tile => ~2^-16 relative error).
//
// Viterbi forward is max-plus (not a matmul): wave-per-batch, lane=state,
// alpha broadcast via LDS, backpointers kept in LDS so the serial backtrack
// is LDS-latency instead of a 511-deep global pointer chase.
//
// Single-wave blocks exploit the ISA guarantee that per-wave LDS ops are
// processed in order: compiler-only fences replace s_barrier.
// ---------------------------------------------------------------------------

#define CRF_B 2048
#define CRF_T 512
#define CRF_N 32
#define CRF_TN (CRF_T * CRF_N)

typedef __attribute__((ext_vector_type(16))) __bf16 v16bf;
typedef __attribute__((ext_vector_type(8)))  float  v8f;
typedef __attribute__((ext_vector_type(8)))  int    v8i;

// compiler-only ordering fence (hardware keeps per-wave LDS ops in order)
#define WAVE_FENCE() asm volatile("" ::: "memory")

// raw hardware transcendentals (domain is denormal-safe here)
__device__ __forceinline__ float exp_n(float x) {
    return __builtin_amdgcn_exp2f(x * 1.44269504088896340736f);
}
__device__ __forceinline__ float log_n(float x) {
    return __builtin_amdgcn_logf(x) * 0.69314718055994530942f;
}

// lane ^= M butterfly via single ds_swizzle (xor mode)
template <int M>
__device__ __forceinline__ float swzf(float x) {
    int r = __builtin_amdgcn_ds_swizzle(__builtin_bit_cast(int, x),
                                        0x1F | (M << 10));
    return __builtin_bit_cast(float, r);
}
template <int M>
__device__ __forceinline__ int swzi(int x) {
    return __builtin_amdgcn_ds_swizzle(x, 0x1F | (M << 10));
}

// f32 -> bf16 RNE (used only for the one-time E matrix build)
__device__ __forceinline__ __bf16 f2bf_rne(float f, float* back) {
    unsigned u = __builtin_bit_cast(unsigned, f);
    unsigned r = (u + 0x7FFFu + ((u >> 16) & 1u)) >> 16;
    *back = __builtin_bit_cast(float, r << 16);
    return __builtin_bit_cast(__bf16, (unsigned short)r);
}

// ===========================================================================
// Kernel A: log-partition (forward algorithm) via WMMA. One wave = 16 batches.
// alpha kept in WMMA C-layout (two 16x16 f32 tiles: states 0-15 / 16-31).
// ===========================================================================
__global__ __launch_bounds__(32)
void crf_lognorm_wmma(const float* __restrict__ logits,
                      const float* __restrict__ trans,
                      const int*   __restrict__ nwords,
                      float*       __restrict__ out_lognorm) {
    const int lane = threadIdx.x;          // 0..31
    const int b0   = blockIdx.x * 16;      // 16 batches per wave
    const int hi   = lane >> 4;            // lane half
    const int lo   = lane & 15;

    // padded transpose buffer: stride 36 -> 8*36 mod 64 == 32, no bank conflicts
    __shared__ __align__(16) float sh[16 * 36];

    // ---- Build E = exp(trans) in WMMA B-layout, bf16 hi/lo split ----------
    // B-layout (32x16 16-bit): lane holds one N column; K in table order.
    const int krow = lo + 16 * hi;
    v16bf EhL, ElL, EhH, ElH;
#pragma unroll
    for (int e = 0; e < 16; ++e) {
        const int k = (e < 8) ? (8 * hi + e) : (8 + 8 * hi + e);
        float xL = exp_n(trans[k * CRF_N + lo]);
        float xH = exp_n(trans[k * CRF_N + 16 + lo]);
        float bL, bH, d;
        EhL[e] = f2bf_rne(xL, &bL);
        ElL[e] = f2bf_rne(xL - bL, &d);
        EhH[e] = f2bf_rne(xH, &bH);
        ElH[e] = f2bf_rne(xH - bH, &d);
    }

    // ---- per-row nwords (C-layout rows) + wave max for loop bound ---------
    int nw[8];
#pragma unroll
    for (int r = 0; r < 8; ++r) nw[r] = nwords[b0 + r + 8 * hi];
    int maxnw = nw[0];
#pragma unroll
    for (int r = 1; r < 8; ++r) maxnw = max(maxnw, nw[r]);
    maxnw = max(maxnw, swzi<16>(maxnw));
    maxnw = max(maxnw, swzi<8>(maxnw));
    maxnw = max(maxnw, swzi<4>(maxnw));
    maxnw = max(maxnw, swzi<2>(maxnw));
    maxnw = max(maxnw, swzi<1>(maxnw));

    // ---- alpha init = logits[:, 0, :] in C-layout -------------------------
    float aL[8], aH[8];
#pragma unroll
    for (int r = 0; r < 8; ++r) {
        const float* row = logits + (size_t)(b0 + r + 8 * hi) * CRF_TN;
        aL[r] = row[lo];
        aH[r] = row[lo + 16];
    }

    const int c0 = hi ? 8 : 0;    // K runs owned by this lane (A-layout)
    const int c1 = hi ? 24 : 16;

    for (int t = 1; t < maxnw; ++t) {
        // this step's emissions (C-layout) + software prefetch 4 steps ahead
        float lL[8], lH[8];
#pragma unroll
        for (int r = 0; r < 8; ++r) {
            const float* row =
                logits + (size_t)(b0 + r + 8 * hi) * CRF_TN + (size_t)t * CRF_N;
            lL[r] = row[lo];
            lH[r] = row[lo + 16];
            __builtin_prefetch(row + 4 * CRF_N + lo, 0, 3);
        }

        // ---- transpose alpha C-layout -> A-layout via LDS -----------------
        WAVE_FENCE();                       // prior reads before overwrite
#pragma unroll
        for (int r = 0; r < 8; ++r) {
            sh[(r + 8 * hi) * 36 + lo]      = aL[r];
            sh[(r + 8 * hi) * 36 + 16 + lo] = aH[r];
        }
        WAVE_FENCE();                       // stores before loads

        float av[16];
        {
            const float4* p0 = (const float4*)&sh[lo * 36 + c0];
            const float4* p1 = (const float4*)&sh[lo * 36 + c1];
            float4 x0 = p0[0], x1 = p0[1], x2 = p1[0], x3 = p1[1];
            av[0] = x0.x; av[1] = x0.y; av[2]  = x0.z; av[3]  = x0.w;
            av[4] = x1.x; av[5] = x1.y; av[6]  = x1.z; av[7]  = x1.w;
            av[8] = x2.x; av[9] = x2.y; av[10] = x2.z; av[11] = x2.w;
            av[12] = x3.x; av[13] = x3.y; av[14] = x3.z; av[15] = x3.w;
        }

        // ---- per-batch max (lanes l and l+16 hold the same batch) ---------
        float m = av[0];
#pragma unroll
        for (int e = 1; e < 16; ++e) m = fmaxf(m, av[e]);
        m = fmaxf(m, swzf<16>(m));

        // ---- p = exp(alpha - m) -------------------------------------------
        float pp[16];
#pragma unroll
        for (int e = 0; e < 16; ++e) pp[e] = exp_n(av[e] - m);

        // ---- bf16 hi/lo split, packed 2-at-a-time with v_perm_b32 ---------
        v8i Ahd, Ald;
#pragma unroll
        for (int v = 0; v < 8; ++v) {
            unsigned u0 = __builtin_bit_cast(unsigned, pp[2 * v]);
            unsigned u1 = __builtin_bit_cast(unsigned, pp[2 * v + 1]);
            Ahd[v] = (int)__builtin_amdgcn_perm(u1, u0, 0x07060302u);
            float h0 = __builtin_bit_cast(float, u0 & 0xFFFF0000u);
            float h1 = __builtin_bit_cast(float, u1 & 0xFFFF0000u);
            unsigned l0 = __builtin_bit_cast(unsigned, pp[2 * v] - h0);
            unsigned l1 = __builtin_bit_cast(unsigned, pp[2 * v + 1] - h1);
            Ald[v] = (int)__builtin_amdgcn_perm(l1, l0, 0x07060302u);
        }
        v16bf Ahi = __builtin_bit_cast(v16bf, Ahd);
        v16bf Alo = __builtin_bit_cast(v16bf, Ald);

        // ---- Q = p @ E  (3-term compensated bf16 WMMA per 16x16 tile) -----
        v8f qL = {};
        qL = __builtin_amdgcn_wmma_f32_16x16x32_bf16(false, Ahi, false, EhL,
                                                     (short)0, qL, false, false);
        qL = __builtin_amdgcn_wmma_f32_16x16x32_bf16(false, Alo, false, EhL,
                                                     (short)0, qL, false, false);
        qL = __builtin_amdgcn_wmma_f32_16x16x32_bf16(false, Ahi, false, ElL,
                                                     (short)0, qL, false, false);
        v8f qH = {};
        qH = __builtin_amdgcn_wmma_f32_16x16x32_bf16(false, Ahi, false, EhH,
                                                     (short)0, qH, false, false);
        qH = __builtin_amdgcn_wmma_f32_16x16x32_bf16(false, Alo, false, EhH,
                                                     (short)0, qH, false, false);
        qH = __builtin_amdgcn_wmma_f32_16x16x32_bf16(false, Ahi, false, ElH,
                                                     (short)0, qH, false, false);

        // ---- broadcast m into C-layout, update alpha with mask ------------
        float mC[8];
#pragma unroll
        for (int r = 0; r < 8; ++r) mC[r] = __shfl(m, r + 8 * hi);

#pragma unroll
        for (int r = 0; r < 8; ++r) {
            float nvL  = mC[r] + log_n(qL[r]) + lL[r];
            float nvH  = mC[r] + log_n(qH[r]) + lH[r];
            bool  keep = (t < nw[r]);
            aL[r] = keep ? nvL : aL[r];
            aH[r] = keep ? nvH : aH[r];
        }
    }

    // ---- final per-batch logsumexp over states ----------------------------
    WAVE_FENCE();
#pragma unroll
    for (int r = 0; r < 8; ++r) {
        sh[(r + 8 * hi) * 36 + lo]      = aL[r];
        sh[(r + 8 * hi) * 36 + 16 + lo] = aH[r];
    }
    WAVE_FENCE();
    float av[16];
    {
        const float4* p0 = (const float4*)&sh[lo * 36 + c0];
        const float4* p1 = (const float4*)&sh[lo * 36 + c1];
        float4 x0 = p0[0], x1 = p0[1], x2 = p1[0], x3 = p1[1];
        av[0] = x0.x; av[1] = x0.y; av[2]  = x0.z; av[3]  = x0.w;
        av[4] = x1.x; av[5] = x1.y; av[6]  = x1.z; av[7]  = x1.w;
        av[8] = x2.x; av[9] = x2.y; av[10] = x2.z; av[11] = x2.w;
        av[12] = x3.x; av[13] = x3.y; av[14] = x3.z; av[15] = x3.w;
    }
    float m = av[0];
#pragma unroll
    for (int e = 1; e < 16; ++e) m = fmaxf(m, av[e]);
    m = fmaxf(m, swzf<16>(m));
    float s = 0.f;
#pragma unroll
    for (int e = 0; e < 16; ++e) s += exp_n(av[e] - m);
    s += swzf<16>(s);
    float ln = m + log_n(s);
    if (lane < 16) out_lognorm[b0 + lane] = ln;
}

// ===========================================================================
// Kernel B: Viterbi forward (max-plus) + LDS backtrack + sequence score.
// One wave per batch, lane = state. Backpointers stay in LDS (16 KB).
// ===========================================================================
__global__ __launch_bounds__(32)
void crf_viterbi(const float* __restrict__ logits,
                 const float* __restrict__ trans,
                 const int*   __restrict__ nwords,
                 const int*   __restrict__ tags,
                 float*       __restrict__ pred_out,
                 float*       __restrict__ seq_out) {
    const int lane = threadIdx.x;
    const int b    = blockIdx.x;

    __shared__ __align__(16) float shv[CRF_N];
    __shared__ unsigned char bp[CRF_T * CRF_N];   // 16 KB backpointers
    __shared__ unsigned char ptag[CRF_T];

    const float* lrow = logits + (size_t)b * CRF_TN;
    const int    nw   = nwords[b];

    // trans column for this lane's state j: tcol[i] = trans[i][j]
    float tcol[CRF_N];
#pragma unroll
    for (int i = 0; i < CRF_N; ++i) tcol[i] = trans[i * CRF_N + lane];

    float v = lrow[lane];                          // t = 0

    for (int t = 1; t < nw; ++t) {
        float logit = lrow[(size_t)t * CRF_N + lane];
        WAVE_FENCE();                              // prior reads done first
        shv[lane] = v;
        WAVE_FENCE();                              // store before loads
        float vv[CRF_N];
        {
            const float4* p = (const float4*)shv;  // broadcast reads
#pragma unroll
            for (int q = 0; q < 8; ++q) {
                float4 x = p[q];
                vv[4 * q]     = x.x; vv[4 * q + 1] = x.y;
                vv[4 * q + 2] = x.z; vv[4 * q + 3] = x.w;
            }
        }
        float best = vv[0] + tcol[0];
        int   bidx = 0;
#pragma unroll
        for (int i = 1; i < CRF_N; ++i) {
            float cand = vv[i] + tcol[i];
            bool  c    = cand > best;               // argmax keeps first max
            best = c ? cand : best;
            bidx = c ? i : bidx;
        }
        v = best + logit;
        bp[t * CRF_N + lane] = (unsigned char)bidx;
    }

    // cross-lane argmax of final v (first-index tiebreak, matches jnp.argmax)
    float mv = v;
    int   mi = lane;
    {
        float ov; int oi; bool take;
        ov = swzf<16>(mv); oi = swzi<16>(mi);
        take = (ov > mv) || ((ov == mv) && (oi < mi));
        mv = take ? ov : mv; mi = take ? oi : mi;
        ov = swzf<8>(mv);  oi = swzi<8>(mi);
        take = (ov > mv) || ((ov == mv) && (oi < mi));
        mv = take ? ov : mv; mi = take ? oi : mi;
        ov = swzf<4>(mv);  oi = swzi<4>(mi);
        take = (ov > mv) || ((ov == mv) && (oi < mi));
        mv = take ? ov : mv; mi = take ? oi : mi;
        ov = swzf<2>(mv);  oi = swzi<2>(mi);
        take = (ov > mv) || ((ov == mv) && (oi < mi));
        mv = take ? ov : mv; mi = take ? oi : mi;
        ov = swzf<1>(mv);  oi = swzi<1>(mi);
        take = (ov > mv) || ((ov == mv) && (oi < mi));
        mv = take ? ov : mv; mi = take ? oi : mi;
    }

    WAVE_FENCE();
    if (lane == 0) {
        int cur = mi;
        ptag[nw - 1] = (unsigned char)cur;
        for (int t = nw - 1; t >= 1; --t) {
            cur = bp[t * CRF_N + cur];
            ptag[t - 1] = (unsigned char)cur;
        }
    }
    WAVE_FENCE();

#pragma unroll
    for (int tt = lane; tt < CRF_T; tt += CRF_N)
        pred_out[(size_t)b * CRF_T + tt] = (tt < nw) ? (float)ptag[tt] : 0.0f;

    // ---- sequence score (unary + transition, masked) ----------------------
    const int* trow = tags + (size_t)b * CRF_T;
    float s = 0.f;
    for (int tt = lane; tt < CRF_T; tt += CRF_N) {
        if (tt < nw) {
            int tg = trow[tt];
            s += lrow[(size_t)tt * CRF_N + tg];
            if (tt >= 1) s += trans[trow[tt - 1] * CRF_N + tg];
        }
    }
    s += swzf<16>(s);
    s += swzf<8>(s);
    s += swzf<4>(s);
    s += swzf<2>(s);
    s += swzf<1>(s);
    if (lane == 0) seq_out[b] = s;
}

// ===========================================================================
// Kernel C: loss = mean(lognorm - seq_score)
// ===========================================================================
__global__ __launch_bounds__(256)
void crf_loss(const float* __restrict__ lognorm,
              const float* __restrict__ seq,
              float* __restrict__ out_loss) {
    __shared__ float red[256];
    const int tid = threadIdx.x;
    float s = 0.f;
    for (int i = tid; i < CRF_B; i += 256) s += lognorm[i] - seq[i];
    red[tid] = s;
    __syncthreads();
    for (int w = 128; w >= 1; w >>= 1) {
        if (tid < w) red[tid] += red[tid + w];
        __syncthreads();
    }
    if (tid == 0) *out_loss = red[0] / (float)CRF_B;
}

// ===========================================================================
extern "C" void kernel_launch(void* const* d_in, const int* in_sizes, int n_in,
                              void* d_out, int out_size, void* d_ws, size_t ws_size,
                              hipStream_t stream) {
    const float* logits = (const float*)d_in[0];   // [B,T,N] f32
    const float* trans  = (const float*)d_in[1];   // [N,N]   f32
    const int*   nwords = (const int*)d_in[2];     // [B]
    const int*   tags   = (const int*)d_in[3];     // [B,T]

    float* out      = (float*)d_out;               // [B*T] preds + [1] loss
    float* ws       = (float*)d_ws;
    float* lognorm  = ws;                          // [B]
    float* seqscore = ws + CRF_B;                  // [B]

    crf_lognorm_wmma<<<CRF_B / 16, 32, 0, stream>>>(logits, trans, nwords,
                                                    lognorm);
    crf_viterbi<<<CRF_B, 32, 0, stream>>>(logits, trans, nwords, tags,
                                          out, seqscore);
    crf_loss<<<1, 256, 0, stream>>>(lognorm, seqscore,
                                    out + (size_t)CRF_B * CRF_T);
}
